// PtrNet1_26972394619303
// MI455X (gfx1250) — compile-verified, matrix operand order checked by reading, not run on hardware
//
#include <hip/hip_runtime.h>

// Problem constants (match reference)
constexpr int kB   = 256;
constexpr int kN   = 256;
constexpr int kDin = 64;
constexpr int kH   = 512;

typedef __attribute__((ext_vector_type(16))) _Float16 v16h;
typedef __attribute__((ext_vector_type(8)))  _Float16 v8h;
typedef __attribute__((ext_vector_type(8)))  float    v8f;
typedef __attribute__((ext_vector_type(4)))  int      i32x4;

#if __has_builtin(__builtin_amdgcn_global_load_async_to_lds_b128) && \
    __has_builtin(__builtin_amdgcn_s_wait_asynccnt)
#define USE_ASYNC_LDS 1
typedef i32x4 __attribute__((address_space(1))) gvec4i;  // global <4 x i32>
typedef i32x4 __attribute__((address_space(3))) lvec4i;  // LDS    <4 x i32>
#else
#define USE_ASYNC_LDS 0
#endif

// ---------------------------------------------------------------------------
// WMMA fragment loads (wave32, V_WMMA_F32_16X16X32_F16 operand layout):
//   lane&15 -> row within the 16-row fragment
//   lane&16 -> K sub-half: lanes 0-15 hold K = k0+{0..7,16..23},
//              lanes 16-31 hold K = k0+{8..15,24..31}
// ---------------------------------------------------------------------------
__device__ __forceinline__ v16h load_frag_f16(const _Float16* __restrict__ base,
                                              int ld, int k0, int lane) {
  const _Float16* p = base + (size_t)(lane & 15) * ld + k0 + ((lane & 16) ? 8 : 0);
  v8h lo = *(const v8h*)p;
  v8h hi = *(const v8h*)(p + 16);
  v16h out;
#pragma unroll
  for (int i = 0; i < 8; ++i) { out[i] = lo[i]; out[8 + i] = hi[i]; }
  return out;
}

__device__ __forceinline__ v16h load_frag_lds(const _Float16* sA, int ld, int k0, int lane) {
  const _Float16* p = sA + (lane & 15) * ld + k0 + ((lane & 16) ? 8 : 0);
  v8h lo = *(const v8h*)p;          // ds_load_b128
  v8h hi = *(const v8h*)(p + 16);   // ds_load_b128
  v16h out;
#pragma unroll
  for (int i = 0; i < 8; ++i) { out[i] = lo[i]; out[8 + i] = hi[i]; }
  return out;
}

// ---------------------------------------------------------------------------
// Big GEMM, K = 512 fixed: C(M,512) = A(M,512) @ W(512,512)^T + bias -> f16.
// Block: 256 threads = 8 waves, tile = 16 rows x 512 cols.
// A stripe staged in LDS via async global->LDS copy (1 read instead of 8).
// W fragments register double-buffered so WMMAs don't drain loadcnt.
// ---------------------------------------------------------------------------
__global__ __launch_bounds__(256)
void gemm_big512(const _Float16* __restrict__ A, const _Float16* __restrict__ W,
                 const float* __restrict__ bias, _Float16* __restrict__ Cf16, int M) {
  constexpr int K   = 512;
  constexpr int LDA = K + 8;                 // pad 16B -> rotate LDS banks per row
  __shared__ _Float16 sA[16 * LDA];

  const int tid   = threadIdx.x;
  const int lane  = tid & 31;
  const int wave  = tid >> 5;
  const int mbase = blockIdx.x << 4;
  const int nwave = wave << 6;

  // ---- Stage A row-block (16 x 512 f16 = 16KB) into LDS --------------------
  {
    const int chunks = 16 * K / 8;           // 1024 x 16B
    for (int i = tid; i < chunks; i += 256) {
      const int row  = i >> 6;               // 64 chunks per row
      const int koff = (i & 63) << 3;
      const _Float16* g = A + (size_t)(mbase + row) * K + koff;
      _Float16* l = &sA[row * LDA + koff];
#if USE_ASYNC_LDS
      __builtin_amdgcn_global_load_async_to_lds_b128(
          (gvec4i*)g, (lvec4i*)l, 0, 0);
#else
      *(v8h*)l = *(const v8h*)g;
#endif
    }
#if USE_ASYNC_LDS
    __builtin_amdgcn_s_wait_asynccnt(0);
#endif
    __syncthreads();
  }

  // ---- Main loop: double-buffered W fragments ------------------------------
  v8f  acc[4] = {};
  v16h bf[2][4];
#pragma unroll
  for (int t = 0; t < 4; ++t)
    bf[0][t] = load_frag_f16(W + (size_t)(nwave + t * 16) * K, K, 0, lane);

#pragma unroll
  for (int kk = 0; kk < K / 32; ++kk) {
    const int cur = kk & 1;
    const int nxt = cur ^ 1;
    if (kk + 1 < K / 32) {
#pragma unroll
      for (int t = 0; t < 4; ++t)
        bf[nxt][t] = load_frag_f16(W + (size_t)(nwave + t * 16) * K, K,
                                   (kk + 1) * 32, lane);
    }
    v16h a = load_frag_lds(sA, LDA, kk * 32, lane);
#pragma unroll
    for (int t = 0; t < 4; ++t)
      acc[t] = __builtin_amdgcn_wmma_f32_16x16x32_f16(
          false, a, false, bf[cur][t], (short)0, acc[t], false, false);
  }

  // ---- Epilogue: D layout lanes 0-15: N=lane, M=r; lanes 16-31: M=r+8 ------
  const int lane15 = lane & 15;
  const int mo     = (lane & 16) ? 8 : 0;
#pragma unroll
  for (int t = 0; t < 4; ++t) {
    const int ncol = nwave + t * 16 + lane15;
    const float bv = bias[ncol];
#pragma unroll
    for (int r = 0; r < 8; ++r) {
      const int m = mbase + mo + r;
      Cf16[(size_t)m * kH + ncol] = (_Float16)(acc[t][r] + bv);
    }
  }
}

// ---------------------------------------------------------------------------
// Generic GEMM (embed K=64, u1 M=256): C(M,512) = A(M,K) @ W(512,K)^T + bias
// ---------------------------------------------------------------------------
__global__ __launch_bounds__(256)
void gemm_f16_n512(const _Float16* __restrict__ A, const _Float16* __restrict__ W,
                   const float* __restrict__ bias, float* __restrict__ Cf32,
                   _Float16* __restrict__ Cf16, int M, int K) {
  const int lane  = threadIdx.x & 31;
  const int wave  = threadIdx.x >> 5;
  const int mbase = blockIdx.x << 4;
  const int nwave = wave << 6;

  v8f acc[4] = {};
  const _Float16* Arow = A + (size_t)mbase * K;
  for (int k0 = 0; k0 < K; k0 += 32) {
    v16h a = load_frag_f16(Arow, K, k0, lane);
#pragma unroll
    for (int t = 0; t < 4; ++t) {
      const _Float16* Wrow = W + (size_t)(nwave + t * 16) * K;
      v16h bfr = load_frag_f16(Wrow, K, k0, lane);
      acc[t] = __builtin_amdgcn_wmma_f32_16x16x32_f16(
          false, a, false, bfr, (short)0, acc[t], false, false);
    }
  }

  const int lane15 = lane & 15;
  const int mo     = (lane & 16) ? 8 : 0;
#pragma unroll
  for (int t = 0; t < 4; ++t) {
    const int ncol = nwave + t * 16 + lane15;
    const float bv = bias ? bias[ncol] : 0.0f;
#pragma unroll
    for (int r = 0; r < 8; ++r) {
      const int m = mbase + mo + r;
      const float v = acc[t][r] + bv;
      if (Cf32) Cf32[(size_t)m * kH + ncol] = v;
      if (Cf16) Cf16[(size_t)m * kH + ncol] = (_Float16)v;
    }
  }
}

// ---------------------------------------------------------------------------
// Elementwise helpers
// ---------------------------------------------------------------------------
__global__ void cvt_f32_f16(const float* __restrict__ src, _Float16* __restrict__ dst, int n) {
  int i = blockIdx.x * blockDim.x + threadIdx.x;
  if (i < n) dst[i] = (_Float16)src[i];
}

__global__ void q_init(const float* __restrict__ dec, _Float16* __restrict__ q, int n) {
  int i = blockIdx.x * blockDim.x + threadIdx.x;
  if (i < n) q[i] = (_Float16)dec[i & (kH - 1)];
}

// ---------------------------------------------------------------------------
// Fused attention step: one block (256 threads = 8 waves) per batch row b.
//   scores: u[n] = sum_h vec[h] * tanh_scale * tanh(u1[b,h] + u2[b,n,h]) + 1e9*mask
//   softmax over n (temperature via inv_temp)
//   glimpse: q_out[b,h] = sum_n u2[b,n,h] * a[n]   (f16 for next GEMM)
//   pointer: probs_out[b,n] = a[n]
// ---------------------------------------------------------------------------
__global__ __launch_bounds__(256)
void attend_kernel(const _Float16* __restrict__ u2, const float* __restrict__ u1,
                   const float* __restrict__ vec, const int* __restrict__ mask,
                   _Float16* __restrict__ q_out, float* __restrict__ probs_out,
                   float tanh_scale, float inv_temp, int is_pointer) {
  __shared__ float s_u1[kH];
  __shared__ float s_vec[kH];
  __shared__ float s_u[kN];
  __shared__ float s_red[256];

  const int b   = blockIdx.x;
  const int tid = threadIdx.x;
  for (int h = tid; h < kH; h += 256) {
    s_u1[h]  = u1[(size_t)b * kH + h];
    s_vec[h] = vec[h];
  }
  __syncthreads();

  const int lane = tid & 31;
  const int wave = tid >> 5;
  const _Float16* u2b = u2 + (size_t)b * kN * kH;

  // Phase 1: wave-per-n tanh reduction over h (coalesced f16 row reads)
  for (int n = wave; n < kN; n += 8) {
    const _Float16* row = u2b + (size_t)n * kH;
    float acc = 0.0f;
    for (int h = lane; h < kH; h += 32) {
      float v = s_u1[h] + (float)row[h];
      acc += s_vec[h] * (tanh_scale * tanhf(v));
    }
#pragma unroll
    for (int off = 16; off > 0; off >>= 1) acc += __shfl_down(acc, off, 32);
    if (lane == 0) s_u[n] = acc + 1.0e9f * (float)mask[b * kN + n];
  }
  __syncthreads();

  // Phase 2: softmax over kN == blockDim (thread tid owns n = tid)
  float uval = s_u[tid] * inv_temp;
  s_red[tid] = uval; __syncthreads();
  for (int off = 128; off > 0; off >>= 1) {
    if (tid < off) s_red[tid] = fmaxf(s_red[tid], s_red[tid + off]);
    __syncthreads();
  }
  const float mx = s_red[0]; __syncthreads();
  const float ex = __expf(uval - mx);
  s_red[tid] = ex; __syncthreads();
  for (int off = 128; off > 0; off >>= 1) {
    if (tid < off) s_red[tid] += s_red[tid + off];
    __syncthreads();
  }
  const float a = ex * (1.0f / s_red[0]); __syncthreads();

  if (is_pointer) {
    probs_out[(size_t)b * kN + tid] = a;
    return;
  }

  // Phase 3: context GEMV, thread owns h = tid and tid+256 (h-coalesced)
  s_u[tid] = a; __syncthreads();
  float acc0 = 0.0f, acc1 = 0.0f;
  const int h0 = tid, h1 = tid + 256;
  for (int n = 0; n < kN; ++n) {
    const float an = s_u[n];
    const _Float16* row = u2b + (size_t)n * kH;
    acc0 += an * (float)row[h0];
    acc1 += an * (float)row[h1];
  }
  q_out[(size_t)b * kH + h0] = (_Float16)acc0;
  q_out[(size_t)b * kH + h1] = (_Float16)acc1;
}

// ---------------------------------------------------------------------------
// Host orchestration
// ---------------------------------------------------------------------------
extern "C" void kernel_launch(void* const* d_in, const int* in_sizes, int n_in,
                              void* d_out, int out_size, void* d_ws, size_t ws_size,
                              hipStream_t stream) {
  (void)in_sizes; (void)n_in; (void)out_size; (void)ws_size;

  const float* x     = (const float*)d_in[0];
  const int*   mask  = (const int*)d_in[1];
  const float* W_emb = (const float*)d_in[2];
  const float* b_emb = (const float*)d_in[3];
  const float* dec   = (const float*)d_in[4];
  const float* Wq    = (const float*)d_in[5];
  const float* bq    = (const float*)d_in[6];
  const float* Wref  = (const float*)d_in[7];
  const float* bref  = (const float*)d_in[8];
  const float* Vec   = (const float*)d_in[9];
  const float* Wq2   = (const float*)d_in[10];
  const float* bq2   = (const float*)d_in[11];
  const float* Wref2 = (const float*)d_in[12];
  const float* bref2 = (const float*)d_in[13];
  const float* Vec2  = (const float*)d_in[14];
  float* out = (float*)d_out;

  // Workspace carve-out (256B aligned)
  char* w = (char*)d_ws;
  auto alloc = [&](size_t bytes) -> char* {
    char* p = w;
    w += (bytes + 255) & ~(size_t)255;
    return p;
  };
  _Float16* x_h     = (_Float16*)alloc((size_t)kB * kN * kDin * 2);
  _Float16* Wemb_h  = (_Float16*)alloc((size_t)kH * kDin * 2);
  _Float16* Wq_h    = (_Float16*)alloc((size_t)kH * kH * 2);
  _Float16* Wref_h  = (_Float16*)alloc((size_t)kH * kH * 2);
  _Float16* Wq2_h   = (_Float16*)alloc((size_t)kH * kH * 2);
  _Float16* Wref2_h = (_Float16*)alloc((size_t)kH * kH * 2);
  _Float16* e_h     = (_Float16*)alloc((size_t)kB * kN * kH * 2);  // embeddings
  _Float16* u2_h    = (_Float16*)alloc((size_t)kB * kN * kH * 2);  // (b,n,h) conv output
  _Float16* q_h     = (_Float16*)alloc((size_t)kB * kH * 2);       // query
  float*    u1      = (float*)   alloc((size_t)kB * kH * 4);       // query proj

  auto cvt = [&](const float* s, _Float16* d, int n) {
    cvt_f32_f16<<<(n + 255) / 256, 256, 0, stream>>>(s, d, n);
  };
  cvt(x,     x_h,     kB * kN * kDin);
  cvt(W_emb, Wemb_h,  kH * kDin);
  cvt(Wq,    Wq_h,    kH * kH);
  cvt(Wref,  Wref_h,  kH * kH);
  cvt(Wq2,   Wq2_h,   kH * kH);
  cvt(Wref2, Wref2_h, kH * kH);

  const int Mbig = kB * kN;

  // e = x @ W_emb^T + b_emb         (65536 x 64) @ (64 x 512)
  gemm_f16_n512<<<Mbig / 16, 256, 0, stream>>>(x_h, Wemb_h, b_emb, nullptr, e_h, Mbig, kDin);
  // query0 = broadcast(dec_input)
  q_init<<<(kB * kH + 255) / 256, 256, 0, stream>>>(dec, q_h, kB * kH);

  // Glimpse phase: u2 is query-independent -> compute once for both iterations
  gemm_big512<<<Mbig / 16, 256, 0, stream>>>(e_h, Wref_h, bref, u2_h, Mbig);
  for (int it = 0; it < 2; ++it) {
    gemm_f16_n512<<<kB / 16, 256, 0, stream>>>(q_h, Wq_h, bq, u1, nullptr, kB, kH);
    attend_kernel<<<kB, 256, 0, stream>>>(u2_h, u1, Vec, mask, q_h, nullptr,
                                          /*tanh_scale=*/1.0f, /*inv_temp=*/1.0f / 1.5f,
                                          /*is_pointer=*/0);
  }

  // Pointer phase (reuse u2 buffer)
  gemm_big512<<<Mbig / 16, 256, 0, stream>>>(e_h, Wref2_h, bref2, u2_h, Mbig);
  gemm_f16_n512<<<kB / 16, 256, 0, stream>>>(q_h, Wq2_h, bq2, u1, nullptr, kB, kH);
  attend_kernel<<<kB, 256, 0, stream>>>(u2_h, u1, Vec2, mask, nullptr, out,
                                        /*tanh_scale=*/10.0f, /*inv_temp=*/1.0f,
                                        /*is_pointer=*/1);
}
